// MultiHeadAttention_56238301774215
// MI455X (gfx1250) — matrix operand order repaired and checked by reading, not compiled, not run on hardware
//
#include <hip/hip_runtime.h>
#include <hip/hip_bf16.h>

// Problem constants (match reference)
#define B_ 2
#define S_ 2048
#define D_ 1024
#define H_ 16
#define DK_ 64

typedef __attribute__((ext_vector_type(16))) __bf16 bf16x16;
typedef __attribute__((ext_vector_type(8)))  float  f32x8;
typedef __attribute__((ext_vector_type(4)))  float  f32x4;  // builtin vec for NT stores

union BF16Frag { bf16x16 v; uint4 q[2]; };

// ---- WMMA fragment loaders (bf16 16x16x32, wave32 layouts per ISA 7.12.2) ----
// A (16x32, MxK): lane holds row M=lane&15; half=lane>>4 selects K sub-ranges:
//   VGPRs0-3 = K[half*8 .. +8), VGPRs4-7 = K[16+half*8 .. +8)
__device__ inline bf16x16 load_a_frag(const __bf16* row, int kk, int half) {
  BF16Frag f;
  f.q[0] = *(const uint4*)(row + kk + half * 8);
  f.q[1] = *(const uint4*)(row + kk + 16 + half * 8);
  return f.v;
}
// Same, converting fp32 source (e.g. LDS-staged probabilities) to bf16
__device__ inline bf16x16 a_frag_from_f32(const float* row, int half) {
  bf16x16 v;
  const float* p0 = row + half * 8;
  const float* p1 = row + 16 + half * 8;
#pragma unroll
  for (int j = 0; j < 8; ++j) { v[j] = (__bf16)p0[j]; v[8 + j] = (__bf16)p1[j]; }
  return v;
}
// B (32x16, KxN): lane holds column N=lane&15; K[kk+half*8 .. +8) then K[kk+16+half*8 .. +8)
__device__ inline bf16x16 load_b_frag(const __bf16* col, int kk, int half) {
  BF16Frag f;
  f.q[0] = *(const uint4*)(col + kk + half * 8);
  f.q[1] = *(const uint4*)(col + kk + 16 + half * 8);
  return f.v;
}

__device__ inline f32x8 wmma_bf16(bf16x16 a, bf16x16 b, f32x8 c) {
  return __builtin_amdgcn_wmma_f32_16x16x32_bf16(false, a, false, b, (short)0, c,
                                                 false, false);
}

// ---- fp32 -> bf16 convert, 4 elems/thread (b128 in, b64 out) ----
__global__ __launch_bounds__(256) void cvt_kernel(const float* __restrict__ in,
                                                  __bf16* __restrict__ out, int n4) {
  int i = blockIdx.x * 256 + threadIdx.x;
  if (i < n4) {
    float4 v = ((const float4*)in)[i];
    union { __bf16 h[4]; uint2 u; } o;
    o.h[0] = (__bf16)v.x; o.h[1] = (__bf16)v.y;
    o.h[2] = (__bf16)v.z; o.h[3] = (__bf16)v.w;
    ((uint2*)out)[i] = o.u;
  }
}

// ---- Projection GEMM: Y = X @ W^T + b, with optional RoPE / layout rewrites ----
// mode 0: out fp32 standard [m][c]               (final O-projection)
// mode 1/2: RoPE, bf16 head layout [b][h][s][dk] (Q / K)
// mode 3: no RoPE, bf16 transposed [b][h][dk][s] (V, DK-major for attn@V B-operand)
__global__ __launch_bounds__(128)
void proj_kernel(const __bf16* __restrict__ X, const __bf16* __restrict__ W,
                 const float* __restrict__ bias, float* __restrict__ outF,
                 __bf16* __restrict__ outH, int mode) {
  const int lane = threadIdx.x & 31;
  const int wave = threadIdx.x >> 5;
  const int mt = blockIdx.x >> 4;                 // 256 M-tiles
  const int nt = ((blockIdx.x & 15) << 2) + wave; // 64 N-tiles
  const int m0 = mt * 16, n0 = nt * 16;
  const int r = lane & 15, half = lane >> 4;

  const __bf16* arow = X + (size_t)(m0 + r) * D_;
  const __bf16* brow = W + (size_t)(n0 + r) * D_;

  // Software-pipelined K loop: next-iteration fragments load under current WMMA
  f32x8 c = {};
  bf16x16 a = load_a_frag(arow, 0, half);
  bf16x16 b = load_b_frag(brow, 0, half);
#pragma unroll 2
  for (int kk = 0; kk < D_ - 32; kk += 32) {
    // branchless wrapped prefetch (always in-row, no compare/branch)
    __builtin_prefetch(arow + ((kk + 256) & (D_ - 1)), 0, 1);
    __builtin_prefetch(brow + ((kk + 256) & (D_ - 1)), 0, 1);
    bf16x16 an = load_a_frag(arow, kk + 32, half);
    bf16x16 bn = load_b_frag(brow, kk + 32, half);
    c = wmma_bf16(a, b, c);
    a = an; b = bn;
  }
  c = wmma_bf16(a, b, c);

  const int n = n0 + r;              // this lane's output column
  const float bv = bias[n];
  const int bI = m0 >> 11;           // batch (tile never straddles batches)
  const int h = n >> 6, dk = n & 63;

  if (mode == 0) {
#pragma unroll
    for (int i = 0; i < 8; ++i) {
      int m = m0 + i + 8 * half;
      outF[(size_t)m * D_ + n] = c[i] + bv;
    }
  } else if (mode == 3) {
    // v_t[b][h][dk][s] : 8 consecutive s per lane -> one 16B store
    __bf16 buf[8];
#pragma unroll
    for (int i = 0; i < 8; ++i) buf[i] = (__bf16)(c[i] + bv);
    const int sBase = (m0 & (S_ - 1)) + 8 * half;
    __bf16* dst = outH + ((size_t)(bI * H_ + h) * DK_ + dk) * S_ + sBase;
    *(uint4*)dst = *(const uint4*)buf;
  } else {
    // RoPE then head layout [b][h][s][dk]
    const float inv_freq = __powf(10000.0f, -(float)(dk & ~1) * (1.0f / 64.0f));
#pragma unroll
    for (int i = 0; i < 8; ++i) {
      int s = (m0 & (S_ - 1)) + i + 8 * half;
      float x = c[i] + bv;
      float p = __shfl_xor(x, 1, 32);          // pair partner (dk ^ 1)
      float sn, cs;
      __sincosf((float)s * inv_freq, &sn, &cs);
      float y = (dk & 1) ? (x * cs + p * sn) : (x * cs - p * sn);
      outH[((size_t)(bI * H_ + h) * S_ + s) * DK_ + dk] = (__bf16)y;
    }
  }
}

// ---- Fused attention: scores + softmax + attn write (once) + attn@V ----
// One wave owns one 16-row q-tile of one (b,h). Three passes over K tiles:
//   pass1: row max; pass2: row sum of exp; pass3: p = exp(s-m)/sum -> global attn
//          (nontemporal: 512MB write-once stream must not dirty L2),
//          stage p tile in LDS (D-layout -> A-layout transpose), attn@V WMMA.
// K/V head tiles stay L2/WGP$-resident across the three passes.
__global__ __launch_bounds__(128)
void attn_fused(const __bf16* __restrict__ qh, const __bf16* __restrict__ kh,
                const __bf16* __restrict__ vt, float* __restrict__ attn,
                __bf16* __restrict__ headBf) {
  // 36-float row stride: 144B keeps b128 alignment; 36*r mod 64 distinct for r<16
  __shared__ float stage[4][16][36];

  const int lane = threadIdx.x & 31;
  const int wave = threadIdx.x >> 5;
  const int w = blockIdx.x * 4 + wave;  // 0 .. 4095
  const int qt = w & 127;               // S/16 = 128
  const int bh = w >> 7;                // b*H + h
  const int r = lane & 15, half = lane >> 4;
  const int q0 = qt * 16;

  const __bf16* qrow = qh + ((size_t)bh * S_ + q0 + r) * DK_;
  const __bf16* kbase = kh + (size_t)bh * S_ * DK_;
  float* attnBase = attn + ((size_t)bh * S_ + q0) * S_;

  const bf16x16 a0 = load_a_frag(qrow, 0, half);
  const bf16x16 a1 = load_a_frag(qrow, 32, half);

  // ---- pass 1: per-row max (pipelined over K tiles) ----
  float rm[8];
#pragma unroll
  for (int i = 0; i < 8; ++i) rm[i] = -3.0e38f;
  {
    bf16x16 b0 = load_b_frag(kbase + (size_t)r * DK_, 0, half);
    bf16x16 b1 = load_b_frag(kbase + (size_t)r * DK_, 32, half);
    for (int kt = 0; kt <= qt; ++kt) {
      const int ktn = (kt < qt) ? kt + 1 : qt;  // clamped -> branchless preload
      const __bf16* krow = kbase + (size_t)(ktn * 16 + r) * DK_;
      bf16x16 nb0 = load_b_frag(krow, 0, half);
      bf16x16 nb1 = load_b_frag(krow, 32, half);
      f32x8 c = {};
      c = wmma_bf16(a0, b0, c);
      c = wmma_bf16(a1, b1, c);
      const int kpos = kt * 16 + r;
#pragma unroll
      for (int i = 0; i < 8; ++i) {
        float v = (kpos > q0 + i + 8 * half) ? -1e30f : c[i] * 0.125f;
        rm[i] = fmaxf(rm[i], v);
      }
      b0 = nb0; b1 = nb1;
    }
  }
#pragma unroll
  for (int i = 0; i < 8; ++i) {      // reduce across the 16-lane half
    for (int m = 1; m <= 8; m <<= 1) rm[i] = fmaxf(rm[i], __shfl_xor(rm[i], m, 32));
  }

  // ---- pass 2: per-row sum of exp (pipelined) ----
  float ps[8];
#pragma unroll
  for (int i = 0; i < 8; ++i) ps[i] = 0.0f;
  {
    bf16x16 b0 = load_b_frag(kbase + (size_t)r * DK_, 0, half);
    bf16x16 b1 = load_b_frag(kbase + (size_t)r * DK_, 32, half);
    for (int kt = 0; kt <= qt; ++kt) {
      const int ktn = (kt < qt) ? kt + 1 : qt;
      const __bf16* krow = kbase + (size_t)(ktn * 16 + r) * DK_;
      bf16x16 nb0 = load_b_frag(krow, 0, half);
      bf16x16 nb1 = load_b_frag(krow, 32, half);
      f32x8 c = {};
      c = wmma_bf16(a0, b0, c);
      c = wmma_bf16(a1, b1, c);
      const int kpos = kt * 16 + r;
#pragma unroll
      for (int i = 0; i < 8; ++i) {
        float v = (kpos > q0 + i + 8 * half) ? -1e30f : c[i] * 0.125f;
        ps[i] += __expf(v - rm[i]);  // masked -> exp underflows to exact 0
      }
      b0 = nb0; b1 = nb1;
    }
  }
  float inv[8];
#pragma unroll
  for (int i = 0; i < 8; ++i) {
    for (int m = 1; m <= 8; m <<= 1) ps[i] += __shfl_xor(ps[i], m, 32);
    inv[i] = 1.0f / ps[i];
  }

  // ---- pass 3: normalized p -> global attn (NT) + LDS stage -> attn@V ----
  const __bf16* vbase = vt + (size_t)bh * DK_ * S_;
  f32x8 cav[4] = {{}, {}, {}, {}};
  const int nchunk = (qt + 2) >> 1;  // ceil((qt+1)/2) chunks of K=32
  for (int j = 0; j < nchunk; ++j) {
#pragma unroll
    for (int sub = 0; sub < 2; ++sub) {
      const int kt = 2 * j + sub;
      if (kt <= qt) {
        const __bf16* krow = kbase + (size_t)(kt * 16 + r) * DK_;
        f32x8 c = {};
        c = wmma_bf16(a0, load_b_frag(krow, 0, half), c);
        c = wmma_bf16(a1, load_b_frag(krow, 32, half), c);
        const int kpos = kt * 16 + r;
#pragma unroll
        for (int i = 0; i < 8; ++i) {
          float v = (kpos > q0 + i + 8 * half) ? -1e30f : c[i] * 0.125f;
          float p = __expf(v - rm[i]) * inv[i];
          // write-once attn output: nontemporal, don't dirty L2
          __builtin_nontemporal_store(p,
              attnBase + (size_t)(i + 8 * half) * S_ + kpos);
          stage[wave][i + 8 * half][sub * 16 + r] = p;        // transpose stage
        }
      } else {
#pragma unroll
        for (int i = 0; i < 8; ++i)
          stage[wave][i + 8 * half][sub * 16 + r] = 0.0f;
      }
    }
    // A-fragment of p from LDS (same-wave DS ops are in order)
    const bf16x16 ap = a_frag_from_f32(&stage[wave][r][0], half);
#pragma unroll
    for (int nt = 0; nt < 4; ++nt) {
      const __bf16* vrow = vbase + (size_t)(nt * 16 + r) * S_;
      cav[nt] = wmma_bf16(ap, load_b_frag(vrow, j * 32, half), cav[nt]);
    }
  }

  // ---- causal zero tail of attn (exact zeros), wave-cooperative, NT ----
  const int kZero = (qt + 1) * 16;
  const f32x4 z4 = {0.f, 0.f, 0.f, 0.f};
  for (int row = 0; row < 16; ++row) {
    float* rp = attnBase + (size_t)row * S_;
    for (int col = kZero + lane * 4; col < S_; col += 128)
      __builtin_nontemporal_store(z4, (f32x4*)(rp + col));
  }

  // ---- write head output [b][s][h*64+dk] bf16 ----
  const int bI = bh >> 4, h = bh & 15;
#pragma unroll
  for (int nt = 0; nt < 4; ++nt) {
    const int n = nt * 16 + r;
#pragma unroll
    for (int i = 0; i < 8; ++i) {
      int s = q0 + i + 8 * half;
      headBf[((size_t)bI * S_ + s) * D_ + h * DK_ + n] = (__bf16)cav[nt][i];
    }
  }
}

extern "C" void kernel_launch(void* const* d_in, const int* in_sizes, int n_in,
                              void* d_out, int out_size, void* d_ws, size_t ws_size,
                              hipStream_t stream) {
  const float* Q  = (const float*)d_in[0];
  const float* K  = (const float*)d_in[1];
  const float* V  = (const float*)d_in[2];
  const float* Wq = (const float*)d_in[3];
  const float* bq = (const float*)d_in[4];
  const float* Wk = (const float*)d_in[5];
  const float* bk = (const float*)d_in[6];
  const float* Wv = (const float*)d_in[7];
  const float* bv = (const float*)d_in[8];
  const float* Wo = (const float*)d_in[9];
  const float* bo = (const float*)d_in[10];
  // d_in[11] = causal mask, applied analytically

  float* outP  = (float*)d_out;
  float* attnP = (float*)d_out + (size_t)B_ * S_ * D_;

  // Workspace carve (bf16 elements)
  const size_t NX = (size_t)B_ * S_ * D_;  // 4,194,304
  const size_t NW = (size_t)D_ * D_;       // 1,048,576
  __bf16* ws = (__bf16*)d_ws;
  __bf16* Qbf  = ws;                 // NX
  __bf16* Kbf  = Qbf + NX;           // NX
  __bf16* Vbf  = Kbf + NX;           // NX
  __bf16* Wqb  = Vbf + NX;           // NW
  __bf16* Wkb  = Wqb + NW;
  __bf16* Wvb  = Wkb + NW;
  __bf16* Wob  = Wvb + NW;
  __bf16* q_h  = Wob + NW;           // NX  [b][h][s][dk]
  __bf16* k_h  = q_h + NX;           // NX  [b][h][s][dk]
  __bf16* v_t  = k_h + NX;           // NX  [b][h][dk][s]
  __bf16* head = v_t + NX;           // NX  [b][s][h*64+dk]

  // 1) fp32 -> bf16 converts (4 elems/thread)
  cvt_kernel<<<(int)(NX / 4 / 256), 256, 0, stream>>>(Q, Qbf, (int)(NX / 4));
  cvt_kernel<<<(int)(NX / 4 / 256), 256, 0, stream>>>(K, Kbf, (int)(NX / 4));
  cvt_kernel<<<(int)(NX / 4 / 256), 256, 0, stream>>>(V, Vbf, (int)(NX / 4));
  cvt_kernel<<<(int)(NW / 4 / 256), 256, 0, stream>>>(Wq, Wqb, (int)(NW / 4));
  cvt_kernel<<<(int)(NW / 4 / 256), 256, 0, stream>>>(Wk, Wkb, (int)(NW / 4));
  cvt_kernel<<<(int)(NW / 4 / 256), 256, 0, stream>>>(Wv, Wvb, (int)(NW / 4));
  cvt_kernel<<<(int)(NW / 4 / 256), 256, 0, stream>>>(Wo, Wob, (int)(NW / 4));

  // 2) Q/K/V projections (+RoPE / transpose)
  const int projBlocks = (B_ * S_ / 16) * (D_ / 16 / 4);  // 256 * 16 = 4096
  proj_kernel<<<projBlocks, 128, 0, stream>>>(Qbf, Wqb, bq, nullptr, q_h, 1);
  proj_kernel<<<projBlocks, 128, 0, stream>>>(Kbf, Wkb, bk, nullptr, k_h, 2);
  proj_kernel<<<projBlocks, 128, 0, stream>>>(Vbf, Wvb, bv, nullptr, v_t, 3);

  // 3) fused scores + softmax + attn(write-once) + attn@V
  attn_fused<<<(B_ * H_ * (S_ / 16)) / 4, 128, 0, stream>>>(q_h, k_h, v_t, attnP,
                                                            head);

  // 4) output projection (fp32)
  proj_kernel<<<projBlocks, 128, 0, stream>>>(head, Wob, bo, outP, nullptr, 0);
}
